// GPT2FrozenStateFusion_74028056313941
// MI455X (gfx1250) — compile-verified, matrix operand order checked by reading, not run on hardware
//
#include <hip/hip_runtime.h>

typedef unsigned short u16;
typedef unsigned int   u32;
typedef unsigned long long u64;

typedef __attribute__((ext_vector_type(16))) __bf16 bf16x16;
typedef __attribute__((ext_vector_type(8)))  float  f32x8;
typedef __attribute__((ext_vector_type(4)))  unsigned int v4u;
typedef __attribute__((ext_vector_type(8)))  int v8i;
typedef __attribute__((ext_vector_type(4)))  int v4i;

#define NLAYER 12
#define NHEAD  12
#define EMB    768
#define E3     2304
#define E4     3072
#define DHEAD  64
#define DSTATE 128
#define VOCAB  60
#define BB     8
#define TT     512
#define MTOK   (BB*TT)   // 4096

#if defined(__has_builtin)
#if __has_builtin(__builtin_amdgcn_tensor_load_to_lds) && __has_builtin(__builtin_amdgcn_s_wait_tensorcnt)
#define HAVE_TDM 1
#endif
#endif
#ifndef HAVE_TDM
#define HAVE_TDM 0
#endif

// ---------- bf16 helpers ----------
__device__ __forceinline__ u16 f2bf(float f) {
  union { float f; u32 u; } v; v.f = f;
  u32 r = v.u + 0x7FFFu + ((v.u >> 16) & 1u);
  return (u16)(r >> 16);
}
__device__ __forceinline__ float gelu_f(float x) {
  return 0.5f * x * (1.0f + tanhf(0.7978845608f * (x + 0.044715f * x * x * x)));
}

union ABv { bf16x16 v; u32 u[8]; };

// K-offset pattern for 16-bit A/B fragments (16x32), CDNA5 ISA 7.12.2
__device__ __forceinline__ int koff_of(int p, int g) {
  return (p < 4) ? (2 * p + 8 * g) : (16 + 2 * (p - 4) + 8 * g);
}
__device__ __forceinline__ void load_frag(ABv& f, const u16* base, int g) {
#pragma unroll
  for (int p = 0; p < 8; ++p) f.u[p] = *(const u32*)(base + koff_of(p, g));
}

#if HAVE_TDM
// Build D# per CDNA5 ISA ch.8 and issue TENSOR_LOAD_TO_LDS for a 2D bf16 tile:
// tile0 x tile1 elements from a row-major tensor with row stride stride0 elems.
__device__ __forceinline__ void tdm_load_tile(u32 lds_off, const u16* gsrc,
                                              u32 td0, u32 td1, u32 tile0, u32 tile1,
                                              u32 stride0) {
  u64 ga = (u64)(size_t)gsrc;
  v4u g0;
  g0[0] = 1u;                               // count=1 (valid user descriptor)
  g0[1] = lds_off;                          // lds_addr
  g0[2] = (u32)ga;                          // global_addr[31:0]
  g0[3] = (u32)(ga >> 32) | (2u << 30);     // global_addr[56:32] | type=2
  v8i g1;
  g1[0] = (int)(1u << 16);                  // data_size=1 -> 2-byte elements
  g1[1] = (int)((td0 & 0xFFFFu) << 16);                       // tensor_dim0[15:0]
  g1[2] = (int)((td0 >> 16) | ((td1 & 0xFFFFu) << 16));       // td0[31:16] | td1[15:0]
  g1[3] = (int)((td1 >> 16) | (tile0 << 16));                 // td1[31:16] | tile_dim0
  g1[4] = (int)(tile1 & 0xFFFFu);                             // tile_dim1 (tile_dim2=0)
  g1[5] = (int)stride0;                                       // tensor_dim0_stride[31:0]
  g1[6] = 0;
  g1[7] = 0;
  v4i gz = {0, 0, 0, 0};
#if defined(__clang_major__) && (__clang_major__ >= 23)
  v8i gz8 = {0, 0, 0, 0, 0, 0, 0, 0};
  __builtin_amdgcn_tensor_load_to_lds(g0, g1, gz, gz, gz8, 0);
#else
  __builtin_amdgcn_tensor_load_to_lds(g0, g1, gz, gz, 0);
#endif
}
#endif

// ---------- weight convert: Wt[l][n][k] = bf16(W[l][k][n]), zero-pad n>=N ----------
__global__ void wconv_k(const float* __restrict__ W, u16* __restrict__ Wt,
                        int K, int N, int Npad, int Lc) {
  size_t idx = (size_t)blockIdx.x * blockDim.x + threadIdx.x;
  size_t tot = (size_t)Lc * Npad * K;
  if (idx >= tot) return;
  int k = (int)(idx % K);
  size_t r2 = idx / K;
  int n = (int)(r2 % Npad);
  int l = (int)(r2 / Npad);
  Wt[idx] = (n < N) ? f2bf(W[(size_t)l * K * N + (size_t)k * N + n]) : (u16)0;
}

// ---------- sequential prefix-state scan ----------
__global__ void scan_k(const int* __restrict__ ids, const int* __restrict__ mul,
                       int* __restrict__ pre) {
  int b = threadIdx.x;
  if (b >= BB) return;
  int s = 0; // ID_ID
  for (int t = 0; t < TT; ++t) {
    pre[b * TT + t] = s;
    int x = ids[b * TT + t];
    s = mul[x * VOCAB + s];
  }
}

// ---------- embedding ----------
__global__ void embed_k(const int* __restrict__ ids, const int* __restrict__ pre,
                        const float* __restrict__ tok, const float* __restrict__ semb,
                        const float* __restrict__ spw, const float* __restrict__ spb,
                        const float* __restrict__ wpe, float* __restrict__ h) {
  int token = blockIdx.x, tid = threadIdx.x;
  int t = token & (TT - 1);
  int id = ids[token], ps = pre[token];
  const float* se = semb + (size_t)ps * DSTATE;
#pragma unroll
  for (int j = 0; j < 3; ++j) {
    int e = tid + j * 256;
    float acc = spb[e];
    for (int k = 0; k < DSTATE; ++k) acc += se[k] * spw[k * EMB + e];
    h[(size_t)token * EMB + e] = tok[(size_t)id * EMB + e] + acc + wpe[(size_t)t * EMB + e];
  }
}

// ---------- layernorm -> bf16 ----------
__global__ void ln_k(const float* __restrict__ x, const float* __restrict__ gw,
                     const float* __restrict__ bw, u16* __restrict__ out) {
  int token = blockIdx.x, tid = threadIdx.x;
  __shared__ float red[256];
  const float* xp = x + (size_t)token * EMB;
  float v0 = xp[tid], v1 = xp[tid + 256], v2 = xp[tid + 512];
  red[tid] = v0 + v1 + v2;
  __syncthreads();
  for (int st = 128; st > 0; st >>= 1) { if (tid < st) red[tid] += red[tid + st]; __syncthreads(); }
  float mean = red[0] * (1.0f / EMB);
  __syncthreads();
  float d0 = v0 - mean, d1 = v1 - mean, d2 = v2 - mean;
  red[tid] = d0 * d0 + d1 * d1 + d2 * d2;
  __syncthreads();
  for (int st = 128; st > 0; st >>= 1) { if (tid < st) red[tid] += red[tid + st]; __syncthreads(); }
  float rstd = rsqrtf(red[0] * (1.0f / EMB) + 1e-5f);
  u16* op = out + (size_t)token * EMB;
  op[tid]       = f2bf(d0 * rstd * gw[tid]       + bw[tid]);
  op[tid + 256] = f2bf(d1 * rstd * gw[tid + 256] + bw[tid + 256]);
  op[tid + 512] = f2bf(d2 * rstd * gw[tid + 512] + bw[tid + 512]);
}

// ---------- TDM-staged, register-blocked WMMA GEMM ----------
// Block = 4 waves = 32 rows x 256 cols of output. A tile (32 x 128) staged in LDS by
// the Tensor Data Mover, double-buffered so DMA of chunk i+1 overlaps WMMA of chunk i.
// N must be a multiple of 256. grid = (N/256, M/32).
#define KC 128
template <bool ADD_BIAS, bool GELU_ACT, bool RESID, bool OUT_BF16>
__global__ void gemm_tdm_k(const u16* __restrict__ A, const u16* __restrict__ Bt,
                           const float* __restrict__ bias,
                           float* __restrict__ outF, u16* __restrict__ outB,
                           int K, int ldo) {
  __shared__ u16 sA[2][32 * KC];
  const int w = threadIdx.x >> 5, lane = threadIdx.x & 31;
  const int mp = blockIdx.y;
  const int ng = blockIdx.x * 4 + w;               // 64-column group
  const int g = lane >> 4, r = lane & 15;
  const u16* Arow = A + (size_t)mp * 32 * K;
  const u16* Bp[4];
#pragma unroll
  for (int j = 0; j < 4; ++j) Bp[j] = Bt + (size_t)(ng * 64 + j * 16 + r) * K;

  const int nch = K / KC;
#if HAVE_TDM
  if (w == 0) {
    tdm_load_tile((u32)(size_t)(void*)&sA[0][0], Arow, (u32)K, 32u, KC, 32u, (u32)K);
    __builtin_amdgcn_s_wait_tensorcnt(0);
  }
#else
  for (int q = (int)threadIdx.x; q < 32 * KC / 2; q += 128) {
    int e = q * 2, row = e / KC, col = e % KC;
    ((u32*)sA[0])[q] = *(const u32*)(Arow + (size_t)row * K + col);
  }
#endif
  __syncthreads();

  f32x8 c[2][4] = {};
  for (int ch = 0; ch < nch; ++ch) {
    u16* cur = sA[ch & 1];
    u16* nxt = sA[(ch & 1) ^ 1];
    if (ch + 1 < nch) {
#if HAVE_TDM
      if (w == 0)
        tdm_load_tile((u32)(size_t)(void*)nxt, Arow + (ch + 1) * KC, (u32)K, 32u, KC, 32u, (u32)K);
#else
      for (int q = (int)threadIdx.x; q < 32 * KC / 2; q += 128) {
        int e = q * 2, row = e / KC, col = e % KC;
        ((u32*)nxt)[q] = *(const u32*)(Arow + (size_t)row * K + (ch + 1) * KC + col);
      }
#endif
    }
    const u16* s0 = cur + r * KC;
    const u16* s1 = cur + (16 + r) * KC;
#pragma unroll
    for (int k0 = 0; k0 < KC; k0 += 32) {
      int kg = ch * KC + k0;
      if (kg + 256 < K) __builtin_prefetch((const void*)(Bp[0] + kg + 256), 0, 1);
      ABv a0, a1, b;
      load_frag(a0, s0 + k0, g);                   // LDS -> ds_load_b128 fragments
      load_frag(a1, s1 + k0, g);
#pragma unroll
      for (int j = 0; j < 4; ++j) {
        load_frag(b, Bp[j] + kg, g);
        c[0][j] = __builtin_amdgcn_wmma_f32_16x16x32_bf16(false, a0.v, false, b.v, (short)0, c[0][j], false, false);
        c[1][j] = __builtin_amdgcn_wmma_f32_16x16x32_bf16(false, a1.v, false, b.v, (short)0, c[1][j], false, false);
      }
    }
#if HAVE_TDM
    if (w == 0) __builtin_amdgcn_s_wait_tensorcnt(0);
#endif
    __syncthreads();
  }
#pragma unroll
  for (int j = 0; j < 4; ++j) {
    int n = ng * 64 + j * 16 + r;
    float bn = ADD_BIAS ? bias[n] : 0.0f;
#pragma unroll
    for (int i = 0; i < 2; ++i) {
#pragma unroll
      for (int rr = 0; rr < 8; ++rr) {
        int m = mp * 32 + i * 16 + rr + 8 * g;
        float v = c[i][j][rr] + bn;
        if (GELU_ACT) v = gelu_f(v);
        size_t idx = (size_t)m * ldo + n;
        if (RESID)         outF[idx] += v;
        else if (OUT_BF16) outB[idx] = f2bf(v);
        else               outF[idx]  = v;
      }
    }
  }
}

// ---------- per-wave WMMA GEMM (used for the small vocab head) ----------
template <bool ADD_BIAS, bool RESID, bool OUT_BF16>
__global__ void gemm_wmma_k(const u16* __restrict__ A, const u16* __restrict__ Bt,
                            const float* __restrict__ bias,
                            float* __restrict__ outF, u16* __restrict__ outB,
                            int Mpairs, int Ngroups, int K, int ldo, int Nvalid) {
  int wid = blockIdx.x * 4 + (threadIdx.x >> 5);
  if (wid >= Mpairs * Ngroups) return;
  int lane = threadIdx.x & 31;
  int mp = wid / Ngroups, ng = wid % Ngroups;
  int g = lane >> 4, r = lane & 15;
  const u16* Ap0 = A + (size_t)(mp * 32 + r) * K;
  const u16* Ap1 = Ap0 + (size_t)16 * K;
  const u16* Bp[4];
#pragma unroll
  for (int j = 0; j < 4; ++j) Bp[j] = Bt + (size_t)(ng * 64 + j * 16 + r) * K;
  f32x8 c[2][4] = {};
  for (int k0 = 0; k0 < K; k0 += 32) {
    ABv a0, a1, b;
    load_frag(a0, Ap0 + k0, g);
    load_frag(a1, Ap1 + k0, g);
#pragma unroll
    for (int j = 0; j < 4; ++j) {
      load_frag(b, Bp[j] + k0, g);
      c[0][j] = __builtin_amdgcn_wmma_f32_16x16x32_bf16(false, a0.v, false, b.v, (short)0, c[0][j], false, false);
      c[1][j] = __builtin_amdgcn_wmma_f32_16x16x32_bf16(false, a1.v, false, b.v, (short)0, c[1][j], false, false);
    }
  }
#pragma unroll
  for (int j = 0; j < 4; ++j) {
    int n = ng * 64 + j * 16 + r;
    if (n < Nvalid) {
      float bn = ADD_BIAS ? bias[n] : 0.0f;
#pragma unroll
      for (int i = 0; i < 2; ++i) {
#pragma unroll
        for (int rr = 0; rr < 8; ++rr) {
          int m = mp * 32 + i * 16 + rr + 8 * g;
          float v = c[i][j][rr] + bn;
          size_t idx = (size_t)m * ldo + n;
          if (RESID)         outF[idx] += v;
          else if (OUT_BF16) outB[idx] = f2bf(v);
          else               outF[idx]  = v;
        }
      }
    }
  }
}

// ---------- attention scores: S = Q K^T * scale, causal masked ----------
__global__ void attn_scores_k(const u16* __restrict__ qkv, float* __restrict__ scores) {
  int lane = threadIdx.x & 31;
  int sg = __builtin_amdgcn_readfirstlane((int)(blockIdx.x * 4 + (threadIdx.x >> 5)));
  int tt = blockIdx.y;
  int bh = blockIdx.z;
  int b = bh / NHEAD, hh = bh % NHEAD;
  int g = lane >> 4, r = lane & 15;
  float* srow = scores + (size_t)bh * TT * TT;
  const u16* Qp = qkv + (size_t)(b * TT + tt * 16 + r) * E3 + hh * DHEAD;
  ABv qa[2];
  load_frag(qa[0], Qp, g);
  load_frag(qa[1], Qp + 32, g);
#pragma unroll
  for (int j = 0; j < 4; ++j) {
    int ts = sg * 4 + j;                          // SGPR: scalar branch below
    int s = ts * 16 + r;
    if (ts > tt) {
#pragma unroll
      for (int rr = 0; rr < 8; ++rr)
        srow[(size_t)(tt * 16 + rr + 8 * g) * TT + s] = -1e9f;
      continue;
    }
    const u16* Kp = qkv + (size_t)(b * TT + s) * E3 + EMB + hh * DHEAD;
    f32x8 cc = {};
#pragma unroll
    for (int kk = 0; kk < 2; ++kk) {
      ABv bm;
      load_frag(bm, Kp + kk * 32, g);
      cc = __builtin_amdgcn_wmma_f32_16x16x32_bf16(false, qa[kk].v, false, bm.v, (short)0, cc, false, false);
    }
#pragma unroll
    for (int rr = 0; rr < 8; ++rr) {
      int t = tt * 16 + rr + 8 * g;
      float v = cc[rr] * 0.125f;
      if (s > t) v = -1e9f;
      srow[(size_t)t * TT + s] = v;
    }
  }
}

// ---------- row softmax (f32 in, bf16 out) ----------
__global__ void softmax_k(const float* __restrict__ sc, u16* __restrict__ pb) {
  size_t row = blockIdx.x;
  int tid = threadIdx.x;
  __shared__ float red[256];
  const float* xp = sc + row * TT;
  float a = xp[tid], b = xp[tid + 256];
  red[tid] = fmaxf(a, b);
  __syncthreads();
  for (int st = 128; st > 0; st >>= 1) { if (tid < st) red[tid] = fmaxf(red[tid], red[tid + st]); __syncthreads(); }
  float mx = red[0];
  __syncthreads();
  float ea = __expf(a - mx), eb = __expf(b - mx);
  red[tid] = ea + eb;
  __syncthreads();
  for (int st = 128; st > 0; st >>= 1) { if (tid < st) red[tid] += red[tid + st]; __syncthreads(); }
  float inv = 1.0f / red[0];
  pb[row * TT + tid]       = f2bf(ea * inv);
  pb[row * TT + tid + 256] = f2bf(eb * inv);
}

// ---------- V transpose: vt[bh][n][s] ----------
__global__ void vtrans_k(const u16* __restrict__ qkv, u16* __restrict__ vt) {
  size_t idx = (size_t)blockIdx.x * blockDim.x + threadIdx.x;
  if (idx >= (size_t)BB * NHEAD * DHEAD * TT) return;
  int s = (int)(idx & (TT - 1));
  size_t r2 = idx >> 9;
  int n = (int)(r2 & (DHEAD - 1));
  int bh = (int)(r2 >> 6);
  int b = bh / NHEAD, hh = bh % NHEAD;
  vt[idx] = qkv[(size_t)(b * TT + s) * E3 + 2 * EMB + hh * DHEAD + n];
}

// ---------- O = P V: wave handles 1 t-tile x 4 d-tiles ----------
__global__ void attn_pv_k(const u16* __restrict__ pb, const u16* __restrict__ vt,
                          u16* __restrict__ o) {
  int lane = threadIdx.x & 31;
  int tt = __builtin_amdgcn_readfirstlane((int)(blockIdx.x * 4 + (threadIdx.x >> 5)));
  int bh = blockIdx.y;
  int b = bh / NHEAD, hh = bh % NHEAD;
  int g = lane >> 4, r = lane & 15;
  const u16* Pp = pb + (size_t)bh * TT * TT + (size_t)(tt * 16 + r) * TT;
  const u16* Vb = vt + (size_t)bh * DHEAD * TT + (size_t)r * TT;
  f32x8 c[4] = {};
  for (int k0 = 0; k0 < TT; k0 += 32) {
    ABv a, bm;
    load_frag(a, Pp + k0, g);
#pragma unroll
    for (int j = 0; j < 4; ++j) {
      load_frag(bm, Vb + (size_t)(j * 16) * TT + k0, g);
      c[j] = __builtin_amdgcn_wmma_f32_16x16x32_bf16(false, a.v, false, bm.v, (short)0, c[j], false, false);
    }
  }
#pragma unroll
  for (int j = 0; j < 4; ++j) {
    int n = j * 16 + r;
#pragma unroll
    for (int rr = 0; rr < 8; ++rr) {
      int t = tt * 16 + rr + 8 * g;
      o[(size_t)(b * TT + t) * EMB + hh * DHEAD + n] = f2bf(c[j][rr]);
    }
  }
}

// =======================================================================
extern "C" void kernel_launch(void* const* d_in, const int* in_sizes, int n_in,
                              void* d_out, int out_size, void* d_ws, size_t ws_size,
                              hipStream_t stream) {
  const int*   input_ids   = (const int*)d_in[0];
  const int*   mul         = (const int*)d_in[1];
  const float* tok_emb     = (const float*)d_in[2];
  const float* state_emb   = (const float*)d_in[3];
  const float* state_projw = (const float*)d_in[4];
  const float* state_projb = (const float*)d_in[5];
  const float* wpe         = (const float*)d_in[6];
  const float* ln1_g       = (const float*)d_in[7];
  const float* ln1_b       = (const float*)d_in[8];
  const float* attn_w      = (const float*)d_in[9];
  const float* attn_b      = (const float*)d_in[10];
  const float* attn_pw     = (const float*)d_in[11];
  const float* attn_pb     = (const float*)d_in[12];
  const float* ln2_g       = (const float*)d_in[13];
  const float* ln2_b       = (const float*)d_in[14];
  const float* fc_w        = (const float*)d_in[15];
  const float* fc_b        = (const float*)d_in[16];
  const float* mlp_pw      = (const float*)d_in[17];
  const float* mlp_pb      = (const float*)d_in[18];
  const float* lnf_g       = (const float*)d_in[19];
  const float* lnf_b       = (const float*)d_in[20];
  const float* head_w      = (const float*)d_in[21];
  const float* head_b      = (const float*)d_in[22];
  float* out = (float*)d_out;

  size_t off = 0;
  auto alloc = [&](size_t bytes) -> char* {
    char* p = (char*)d_ws + off;
    off = (off + bytes + 255) & ~(size_t)255;
    return p;
  };
  u16* wt_attn = (u16*)alloc((size_t)NLAYER * E3  * EMB * 2);
  u16* wt_proj = (u16*)alloc((size_t)NLAYER * EMB * EMB * 2);
  u16* wt_fc   = (u16*)alloc((size_t)NLAYER * E4  * EMB * 2);
  u16* wt_mlp  = (u16*)alloc((size_t)NLAYER * EMB * E4  * 2);
  u16* wt_head = (u16*)alloc((size_t)64 * EMB * 2);
  int*   pre   = (int*)  alloc((size_t)MTOK * 4);
  float* h     = (float*)alloc((size_t)MTOK * EMB * 4);
  u16*   abuf  = (u16*)  alloc((size_t)MTOK * EMB * 2);
  u16*   qkv   = (u16*)  alloc((size_t)MTOK * E3  * 2);
  u16*   obuf  = (u16*)  alloc((size_t)MTOK * EMB * 2);
  u16*   mid   = (u16*)  alloc((size_t)MTOK * E4  * 2);
  u16*   vt    = (u16*)  alloc((size_t)BB * NHEAD * DHEAD * TT * 2);
  float* sc    = (float*)alloc((size_t)BB * NHEAD * TT * TT * 4);
  u16*   pbuf  = (u16*)  alloc((size_t)BB * NHEAD * TT * TT * 2);

  auto cdiv = [](size_t a, size_t b) { return (int)((a + b - 1) / b); };

  // weight conversion (f32 -> transposed bf16)
  wconv_k<<<cdiv((size_t)NLAYER * E3 * EMB, 256), 256, 0, stream>>>(attn_w, wt_attn, EMB, E3, E3, NLAYER);
  wconv_k<<<cdiv((size_t)NLAYER * EMB * EMB, 256), 256, 0, stream>>>(attn_pw, wt_proj, EMB, EMB, EMB, NLAYER);
  wconv_k<<<cdiv((size_t)NLAYER * E4 * EMB, 256), 256, 0, stream>>>(fc_w, wt_fc, EMB, E4, E4, NLAYER);
  wconv_k<<<cdiv((size_t)NLAYER * EMB * E4, 256), 256, 0, stream>>>(mlp_pw, wt_mlp, E4, EMB, EMB, NLAYER);
  wconv_k<<<cdiv((size_t)64 * EMB, 256), 256, 0, stream>>>(head_w, wt_head, EMB, VOCAB, 64, 1);

  scan_k<<<1, 32, 0, stream>>>(input_ids, mul, pre);
  embed_k<<<MTOK, 256, 0, stream>>>(input_ids, pre, tok_emb, state_emb, state_projw, state_projb, wpe, h);

  const int Mpairs = MTOK / 32;  // 128
  for (int l = 0; l < NLAYER; ++l) {
    ln_k<<<MTOK, 256, 0, stream>>>(h, ln1_g + l * EMB, ln1_b + l * EMB, abuf);
    gemm_tdm_k<true, false, false, true><<<dim3(E3 / 256, Mpairs), 128, 0, stream>>>(
        abuf, wt_attn + (size_t)l * E3 * EMB, attn_b + (size_t)l * E3,
        nullptr, qkv, EMB, E3);
    attn_scores_k<<<dim3(2, 32, BB * NHEAD), 128, 0, stream>>>(qkv, sc);
    softmax_k<<<BB * NHEAD * TT, 256, 0, stream>>>(sc, pbuf);
    vtrans_k<<<cdiv((size_t)BB * NHEAD * DHEAD * TT, 256), 256, 0, stream>>>(qkv, vt);
    attn_pv_k<<<dim3(8, BB * NHEAD), 128, 0, stream>>>(pbuf, vt, obuf);
    gemm_tdm_k<true, false, true, false><<<dim3(EMB / 256, Mpairs), 128, 0, stream>>>(
        obuf, wt_proj + (size_t)l * EMB * EMB, attn_pb + (size_t)l * EMB,
        h, nullptr, EMB, EMB);
    ln_k<<<MTOK, 256, 0, stream>>>(h, ln2_g + l * EMB, ln2_b + l * EMB, abuf);
    gemm_tdm_k<true, true, false, true><<<dim3(E4 / 256, Mpairs), 128, 0, stream>>>(
        abuf, wt_fc + (size_t)l * E4 * EMB, fc_b + (size_t)l * E4,
        nullptr, mid, EMB, E4);
    gemm_tdm_k<true, false, true, false><<<dim3(EMB / 256, Mpairs), 128, 0, stream>>>(
        mid, wt_mlp + (size_t)l * EMB * E4, mlp_pb + (size_t)l * EMB,
        h, nullptr, E4, EMB);
  }

  ln_k<<<MTOK, 256, 0, stream>>>(h, lnf_g, lnf_b, abuf);
  gemm_wmma_k<true, false, false><<<cdiv((size_t)Mpairs, 4), 128, 0, stream>>>(
      abuf, wt_head, head_b, out, nullptr, Mpairs, 1, EMB, VOCAB, VOCAB);
}
